// TissueGraphNetwork_26946624815831
// MI455X (gfx1250) — matrix operand order repaired and checked by reading, not compiled, not run on hardware
//
#include <hip/hip_runtime.h>
#include <hip/hip_bf16.h>

// ---- problem constants (from reference) ----
#define NODES   8192
#define EDGES   131072
#define DIN     512
#define HDIM    256
#define H2      128
#define NGRAPH  8
#define NCLASS  3
#define NBMAX   1024
#define BPAD    520      // 512 max K + 8 bf16 pad -> 1040B row stride, conflict-free ds_load_b128

typedef __attribute__((ext_vector_type(16))) __bf16 v16bf;
typedef __attribute__((ext_vector_type(8)))  __bf16 v8bf;
typedef __attribute__((ext_vector_type(8)))  float  v8f;

static __device__ __forceinline__ __bf16 f2bf(float f) {
    unsigned u = __builtin_bit_cast(unsigned, f);
    u += 0x7FFFu + ((u >> 16) & 1u);               // round-to-nearest-even
    unsigned short h = (unsigned short)(u >> 16);
    return __builtin_bit_cast(__bf16, h);
}

// ---------------- utility kernels ----------------
__global__ void zero_u32(unsigned* __restrict__ p, int n) {
    int i = blockIdx.x * blockDim.x + threadIdx.x;
    if (i < n) p[i] = 0u;
}

__global__ void cvt_bf16(const float* __restrict__ s, __bf16* __restrict__ d, int n) {
    int i = blockIdx.x * blockDim.x + threadIdx.x;
    if (i < n) d[i] = f2bf(s[i]);
}

// src [R,C] f32 row-major -> dst [C,R] bf16 row-major (B transposed for WMMA B-fragment loads)
__global__ void transpose_cvt(const float* __restrict__ s, __bf16* __restrict__ d, int R, int C) {
    int i = blockIdx.x * blockDim.x + threadIdx.x;
    if (i >= R * C) return;
    int r = i / C, c = i - r * C;
    d[(size_t)c * R + r] = f2bf(s[i]);
}

// adjacency bitmask: bit (u,v) and (v,u), plus the diagonal
__global__ void build_adj(const int* __restrict__ ei, unsigned* __restrict__ mask, int E, int Nn) {
    int i = blockIdx.x * blockDim.x + threadIdx.x;
    const int words = Nn / 32;
    if (i < E) {
        int u = ei[i];
        int v = ei[E + i];
        atomicOr(&mask[(size_t)u * words + (v >> 5)], 1u << (v & 31));
        atomicOr(&mask[(size_t)v * words + (u >> 5)], 1u << (u & 31));
    }
    if (i < Nn) {
        atomicOr(&mask[(size_t)i * words + (i >> 5)], 1u << (i & 31));
    }
}

// ---------------- WMMA GEMM: C[M,N] = A[M,K] (bf16) @ B (Bt[N,K] bf16) + bias ----------------
// block = 256 threads = 8 waves arranged 4(m) x 2(n); each wave owns a 16x16 C tile -> 64x32 tile.
// The 32-row Bt panel for this block is staged in LDS via GLOBAL_LOAD_ASYNC_TO_LDS_B128
// (ASYNCcnt path) so all 4 m-waves share one copy; A streams direct from global.
// Grids are exact multiples (64 x 32): no bounds guard (block barrier needs all threads).
__global__ void wmma_gemm(const __bf16* __restrict__ A, const __bf16* __restrict__ Bt,
                          const float* __restrict__ bias, float* __restrict__ C,
                          int M, int N, int K) {
    __shared__ __bf16 Bsh[32][BPAD];

    // ---- stage the B panel (32 rows x K bf16) into LDS asynchronously ----
    // NOTE: the LDS-address asm operand is derived from the real address of Bsh via
    // ptrtoint (low 32 bits of a flat LDS pointer == wave-relative LDS byte address,
    // ISA 10.2 aperture rules). The ptrtoint capture forces LLVM to assume the
    // "memory"-clobbering asm writes Bsh, so the later ds_loads are not folded away.
    {
        const __bf16* Bpanel = Bt + (size_t)(blockIdx.y * 32) * K;
        const int cpr = K >> 3;              // 16-byte chunks per row (power of two)
        const int shift = 31 - __clz(cpr);
        const int total = 32 * cpr;
        for (int c = threadIdx.x; c < total; c += 256) {
            int r = c >> shift, q = c & (cpr - 1);
            unsigned lofs = (unsigned)(size_t)&Bsh[r][q * 8];          // LDS byte address
            const void* g = (const char*)(Bpanel + (size_t)r * K) + q * 16;
            asm volatile("global_load_async_to_lds_b128 %0, %1, off"
                         :: "v"(lofs), "v"(g) : "memory");
        }
        asm volatile("s_wait_asynccnt 0" ::: "memory");
        __syncthreads();
    }

    const int wave = threadIdx.x >> 5;
    const int lane = threadIdx.x & 31;
    const int wm = wave & 3, wn = wave >> 2;
    const int row0 = blockIdx.x * 64 + wm * 16;
    const int col0 = blockIdx.y * 32 + wn * 16;

    const int l15  = lane & 15;
    const int half = lane >> 4;
    // A fragment (16x32 bf16): lanes 0-15 hold K 0..7 & 16..23; lanes 16-31 hold K 8..15 & 24..31
    const __bf16* arow = A + (size_t)(row0 + l15) * K + half * 8;
    // B fragment (32x16 bf16): lanes 0-15 hold K 0..7 & 16..23; lanes 16-31 hold K 8..15 & 24..31 (matches A)
    const int bl = wn * 16 + l15;            // local panel row

    v8f acc = {};
    for (int kk = 0; kk < K; kk += 32) {
        v8bf alo = *reinterpret_cast<const v8bf*>(arow + kk);
        v8bf ahi = *reinterpret_cast<const v8bf*>(arow + kk + 16);
        v8bf blo = *reinterpret_cast<const v8bf*>(&Bsh[bl][kk + half * 8]);
        v8bf bhi = *reinterpret_cast<const v8bf*>(&Bsh[bl][kk + 16 + half * 8]);
        v16bf a = __builtin_shufflevector(alo, ahi, 0,1,2,3,4,5,6,7,8,9,10,11,12,13,14,15);
        v16bf b = __builtin_shufflevector(blo, bhi, 0,1,2,3,4,5,6,7,8,9,10,11,12,13,14,15);
        acc = __builtin_amdgcn_wmma_f32_16x16x32_bf16(false, a, false, b, (short)0, acc, false, false);
    }
    const float badd = bias ? bias[col0 + l15] : 0.0f;
#pragma unroll
    for (int r = 0; r < 8; r++) {
        int m = row0 + half * 8 + r;         // C layout: vgpr r -> M = (lane<16 ? r : 8+r)
        C[(size_t)m * N + col0 + l15] = acc[r] + badd;
    }
}

// ---------------- encoder epilogue: ReLU then BN; emit f32 X and bf16 X ----------------
__global__ void enc_epilogue(const float* __restrict__ C, const float* __restrict__ mean,
                             const float* __restrict__ var, const float* __restrict__ gamma,
                             const float* __restrict__ beta, float* __restrict__ X,
                             __bf16* __restrict__ Xbf) {
    int i = blockIdx.x * blockDim.x + threadIdx.x;
    if (i >= NODES * HDIM) return;
    int h = i & (HDIM - 1);
    float v = C[i];
    v = v > 0.f ? v : 0.f;
    v = (v - mean[h]) * rsqrtf(var[h] + 1e-5f) * gamma[h] + beta[h];
    X[i] = v;
    Xbf[i] = f2bf(v);
}

// ---------------- GAT scores: s1 = Wh@a1, s2 = Wh@a2 (one wave per row) ----------------
__global__ void gat_scores(const float* __restrict__ Wh, const float* __restrict__ a,
                           float* __restrict__ s1, float* __restrict__ s2) {
    int wave = threadIdx.x >> 5, lane = threadIdx.x & 31;
    int row = blockIdx.x * 8 + wave;
    float v1 = 0.f, v2 = 0.f;
    for (int c = lane; c < HDIM; c += 32) {
        float wv = Wh[(size_t)row * HDIM + c];
        v1 += wv * a[c];
        v2 += wv * a[HDIM + c];
    }
    for (int off = 16; off > 0; off >>= 1) {
        v1 += __shfl_xor(v1, off, 32);
        v2 += __shfl_xor(v2, off, 32);
    }
    if (lane == 0) { s1[row] = v1; s2[row] = v2; }
}

// ---------------- sparse attention + aggregation: one block (256 thr) per row ----------------
// neighbor extraction via deterministic prefix-sum (sorted j order -> bitwise-deterministic sums)
__global__ void gat_attention(const unsigned* __restrict__ adj, const float* __restrict__ Wh,
                              const float* __restrict__ s1, const float* __restrict__ s2,
                              float* __restrict__ X, __bf16* __restrict__ Xbf) {
    __shared__ int   nbr[NBMAX];
    __shared__ float wgt[NBMAX];
    __shared__ float red[256];
    __shared__ int   scan[256];
    const int row = blockIdx.x;
    const int t = threadIdx.x;

    unsigned w = adj[(size_t)row * (NODES / 32) + t];
    int pc = __popc(w);
    scan[t] = pc;
    __syncthreads();
    for (int st = 1; st < 256; st <<= 1) {         // Hillis-Steele inclusive scan
        int v = (t >= st) ? scan[t - st] : 0;
        __syncthreads();
        if (t >= st) scan[t] += v;
        __syncthreads();
    }
    int base = scan[t] - pc;
    int cnt  = scan[255];
    int c = cnt < NBMAX ? cnt : NBMAX;
    {
        unsigned ww = w;
        int p = base;
        while (ww) {
            int b = __ffs(ww) - 1;
            ww &= ww - 1;
            if (p < NBMAX) nbr[p] = t * 32 + b;
            p++;
        }
    }
    __syncthreads();

    const float s1r = s1[row];
    // pass 1: row max of leaky_relu(s1_i + s2_j)
    float lmax = -3.4e38f;
    for (int p = t; p < c; p += 256) {
        float e = s1r + s2[nbr[p]];
        e = e > 0.f ? e : 0.2f * e;
        lmax = fmaxf(lmax, e);
    }
    red[t] = lmax; __syncthreads();
    for (int s = 128; s > 0; s >>= 1) { if (t < s) red[t] = fmaxf(red[t], red[t + s]); __syncthreads(); }
    float rmax = red[0]; __syncthreads();
    // pass 2: exp and sum
    float lsum = 0.f;
    for (int p = t; p < c; p += 256) {
        float e = s1r + s2[nbr[p]];
        e = e > 0.f ? e : 0.2f * e;
        float ex = __expf(e - rmax);
        wgt[p] = ex;
        lsum += ex;
    }
    red[t] = lsum; __syncthreads();
    for (int s = 128; s > 0; s >>= 1) { if (t < s) red[t] += red[t + s]; __syncthreads(); }
    float inv = 1.f / red[0]; __syncthreads();
    // pass 3: thread t owns output column h=t; gather Wh rows of neighbors
    float acc = 0.f;
    for (int p = 0; p < c; p++)
        acc += wgt[p] * inv * Wh[(size_t)nbr[p] * HDIM + t];
    float o = acc > 0.f ? acc : (__expf(acc) - 1.f);   // ELU
    size_t idx = (size_t)row * HDIM + t;
    float xv = X[idx] + o;                              // residual
    X[idx] = xv;
    Xbf[idx] = f2bf(xv);
}

// ---------------- pooling score: s = tanh(T + bp1) @ Wp2 + bp2 (one wave per row) ----------------
__global__ void pool_score(const float* __restrict__ T, const float* __restrict__ bp1,
                           const float* __restrict__ Wp2, const float* __restrict__ bp2,
                           float* __restrict__ s) {
    int wave = threadIdx.x >> 5, lane = threadIdx.x & 31;
    int row = blockIdx.x * 8 + wave;
    float v = 0.f;
    for (int c = lane; c < H2; c += 32) {
        float tv = tanhf(T[(size_t)row * H2 + c] + bp1[c]);
        v += tv * Wp2[c];
    }
    for (int off = 16; off > 0; off >>= 1) v += __shfl_xor(v, off, 32);
    if (lane == 0) s[row] = v + bp2[0];
}

__global__ void seg_max(const float* __restrict__ s, const int* __restrict__ batch, float* __restrict__ m) {
    __shared__ float red[256];
    int g = blockIdx.x, t = threadIdx.x;
    float lmax = -3.4e38f;
    for (int i = t; i < NODES; i += 256) if (batch[i] == g) lmax = fmaxf(lmax, s[i]);
    red[t] = lmax; __syncthreads();
    for (int st = 128; st > 0; st >>= 1) { if (t < st) red[t] = fmaxf(red[t], red[t + st]); __syncthreads(); }
    if (t == 0) m[g] = red[0];
}

__global__ void seg_sum(const float* __restrict__ s, const int* __restrict__ batch,
                        const float* __restrict__ m, float* __restrict__ denom) {
    __shared__ float red[256];
    int g = blockIdx.x, t = threadIdx.x;
    float mg = m[g];
    float ls = 0.f;
    for (int i = t; i < NODES; i += 256) if (batch[i] == g) ls += __expf(s[i] - mg);
    red[t] = ls; __syncthreads();
    for (int st = 128; st > 0; st >>= 1) { if (t < st) red[t] += red[t + st]; __syncthreads(); }
    if (t == 0) denom[g] = red[0];
}

__global__ void pool_weighted(const float* __restrict__ X, const float* __restrict__ s,
                              const int* __restrict__ batch, const float* __restrict__ m,
                              const float* __restrict__ denom, float* __restrict__ g_out) {
    int g = blockIdx.x, h = threadIdx.x;    // 256 threads == HDIM
    float mg = m[g], inv = 1.f / denom[g];
    float acc = 0.f;
    for (int i = 0; i < NODES; i++) {
        if (batch[i] == g)
            acc += __expf(s[i] - mg) * inv * X[(size_t)i * HDIM + h];
    }
    g_out[g * HDIM + h] = acc;
}

__global__ void classifier(const float* __restrict__ gp, const float* __restrict__ Wc1,
                           const float* __restrict__ bc1, const float* __restrict__ Wc2,
                           const float* __restrict__ bc2, float* __restrict__ out) {
    __shared__ float hid[H2];
    int g = blockIdx.x, t = threadIdx.x;    // 128 threads
    float acc = bc1[t];
    for (int h = 0; h < HDIM; h++) acc += gp[g * HDIM + h] * Wc1[h * H2 + t];
    hid[t] = acc > 0.f ? acc : 0.f;
    __syncthreads();
    if (t < NCLASS) {
        float o = bc2[t];
        for (int k = 0; k < H2; k++) o += hid[k] * Wc2[k * NCLASS + t];
        out[g * NCLASS + t] = o;
    }
}

// ---------------- launch ----------------
extern "C" void kernel_launch(void* const* d_in, const int* in_sizes, int n_in,
                              void* d_out, int out_size, void* d_ws, size_t ws_size,
                              hipStream_t stream) {
    (void)in_sizes; (void)n_in; (void)out_size; (void)ws_size;
    const float* node_features = (const float*)d_in[0];
    const int*   edge_index    = (const int*)  d_in[1];
    /* d_in[2] edge_features: unused by reference */
    const int*   batch         = (const int*)  d_in[3];
    const float* W_enc         = (const float*)d_in[4];
    const float* b_enc         = (const float*)d_in[5];
    const float* bn_gamma      = (const float*)d_in[6];
    const float* bn_beta       = (const float*)d_in[7];
    const float* bn_mean       = (const float*)d_in[8];
    const float* bn_var        = (const float*)d_in[9];
    const float* gat_W         = (const float*)d_in[10];
    const float* gat_a         = (const float*)d_in[11];
    const float* Wp1           = (const float*)d_in[12];
    const float* bp1           = (const float*)d_in[13];
    const float* Wp2           = (const float*)d_in[14];
    const float* bp2           = (const float*)d_in[15];
    const float* Wc1           = (const float*)d_in[16];
    const float* bc1           = (const float*)d_in[17];
    const float* Wc2           = (const float*)d_in[18];
    const float* bc2           = (const float*)d_in[19];
    float* out = (float*)d_out;

    // workspace carve-up (256B aligned)
    char* ws = (char*)d_ws;
    size_t off = 0;
    auto alloc = [&](size_t bytes) -> char* {
        char* p = ws + off;
        off += (bytes + 255) & ~(size_t)255;
        return p;
    };
    __bf16* NFbf  = (__bf16*)alloc((size_t)NODES * DIN * 2);       // 8 MB
    __bf16* WencT = (__bf16*)alloc((size_t)HDIM * DIN * 2);        // 256 KB
    __bf16* gatWT = (__bf16*)alloc((size_t)3 * HDIM * HDIM * 2);   // 384 KB
    __bf16* Wp1T  = (__bf16*)alloc((size_t)H2 * HDIM * 2);         // 64 KB
    float*  Cbuf  = (float*) alloc((size_t)NODES * HDIM * 4);      // 8 MB (enc out / Wh / T)
    float*  Xf    = (float*) alloc((size_t)NODES * HDIM * 4);      // 8 MB
    __bf16* Xbf   = (__bf16*)alloc((size_t)NODES * HDIM * 2);      // 4 MB
    unsigned* adj = (unsigned*)alloc((size_t)NODES * (NODES / 32) * 4); // 8 MB
    float*  s1    = (float*) alloc((size_t)NODES * 4);
    float*  s2    = (float*) alloc((size_t)NODES * 4);
    float*  sbuf  = (float*) alloc((size_t)NODES * 4);
    float*  mg    = (float*) alloc((size_t)NGRAPH * 4);
    float*  denom = (float*) alloc((size_t)NGRAPH * 4);
    float*  gpool = (float*) alloc((size_t)NGRAPH * HDIM * 4);

    // 1) adjacency bitmask
    {
        int words = NODES * (NODES / 32);
        zero_u32<<<(words + 255) / 256, 256, 0, stream>>>(adj, words);
        int n = (EDGES > NODES ? EDGES : NODES);
        build_adj<<<(n + 255) / 256, 256, 0, stream>>>(edge_index, adj, EDGES, NODES);
    }
    // 2) precision conversions (weights transposed for WMMA B fragments)
    cvt_bf16<<<(NODES * DIN + 255) / 256, 256, 0, stream>>>(node_features, NFbf, NODES * DIN);
    transpose_cvt<<<(DIN * HDIM + 255) / 256, 256, 0, stream>>>(W_enc, WencT, DIN, HDIM);
    for (int l = 0; l < 3; l++)
        transpose_cvt<<<(HDIM * HDIM + 255) / 256, 256, 0, stream>>>(gat_W + (size_t)l * HDIM * HDIM,
                                                                     gatWT + (size_t)l * HDIM * HDIM,
                                                                     HDIM, HDIM);
    transpose_cvt<<<(HDIM * H2 + 255) / 256, 256, 0, stream>>>(Wp1, Wp1T, HDIM, H2);

    // 3) encoder GEMM (WMMA bf16, async-LDS B panel) + ReLU/BN epilogue
    wmma_gemm<<<dim3(NODES / 64, HDIM / 32), 256, 0, stream>>>(NFbf, WencT, b_enc, Cbuf, NODES, HDIM, DIN);
    enc_epilogue<<<(NODES * HDIM + 255) / 256, 256, 0, stream>>>(Cbuf, bn_mean, bn_var, bn_gamma, bn_beta, Xf, Xbf);

    // 4) three GAT layers: Wh GEMM (WMMA) -> scores -> sparse softmax/aggregate
    for (int l = 0; l < 3; l++) {
        wmma_gemm<<<dim3(NODES / 64, HDIM / 32), 256, 0, stream>>>(Xbf, gatWT + (size_t)l * HDIM * HDIM,
                                                                   nullptr, Cbuf, NODES, HDIM, HDIM);
        gat_scores<<<NODES / 8, 256, 0, stream>>>(Cbuf, gat_a + (size_t)l * 2 * HDIM, s1, s2);
        gat_attention<<<NODES, 256, 0, stream>>>(adj, Cbuf, s1, s2, Xf, Xbf);
    }

    // 5) attention pooling
    wmma_gemm<<<dim3(NODES / 64, H2 / 32), 256, 0, stream>>>(Xbf, Wp1T, nullptr, Cbuf, NODES, H2, HDIM);
    pool_score<<<NODES / 8, 256, 0, stream>>>(Cbuf, bp1, Wp2, bp2, sbuf);
    seg_max<<<NGRAPH, 256, 0, stream>>>(sbuf, batch, mg);
    seg_sum<<<NGRAPH, 256, 0, stream>>>(sbuf, batch, mg, denom);
    pool_weighted<<<NGRAPH, 256, 0, stream>>>(Xf, sbuf, batch, mg, denom, gpool);

    // 6) classifier head
    classifier<<<NGRAPH, H2, 0, stream>>>(gpool, Wc1, bc1, Wc2, bc2, out);
}
